// PointerNet_GNN_8186207666622
// MI455X (gfx1250) — compile-verified, matrix operand order checked by reading, not run on hardware
//
#include <hip/hip_runtime.h>
#include <hip/hip_bf16.h>
#include <math.h>

// ---------------------------------------------------------------------------
// PointerNet-GNN on gfx1250: WMMA f16->f32 GEMMs for GNN + LSTM gate matmuls.
// Encoder BiLSTM is a single persistent kernel over all 512 timesteps:
//   - h state in LDS (barrier-synced per step), c state in accumulator VGPRs,
//   - gate weights hoisted into 320 registers/wave (loop-invariant B frags).
// ---------------------------------------------------------------------------

typedef _Float16 f16;
typedef __attribute__((ext_vector_type(16))) _Float16 v16h;
typedef __attribute__((ext_vector_type(8)))  float    v8f;

#define B_   256
#define M_   512
#define D_   64
#define GH_  256
#define GO_  64
#define H_   256
#define NSEL 32
#define NEGV (-1e9f)

// ---- WMMA fragment helpers (16x16x32 f16, wave32) -------------------------
// 16-bit A-matrix 16x32: lanes 0-15 row M=lane hold K={0..7,16..23} pairs,
// lanes 16-31 hold K={8..15,24..31} (ISA 7.12.2).
__device__ __forceinline__ int kmap(int v, int half) {
  return (v < 4 ? 2 * v : 16 + 2 * (v - 4)) + half * 8;
}

__device__ __forceinline__ v16h load_a16(const f16* p, int ld) {
  int lane = threadIdx.x & 31;
  int row = lane & 15, half = lane >> 4;
  const f16* r = p + (size_t)row * ld;
  v16h a;
#pragma unroll
  for (int v = 0; v < 8; ++v) {
    int k = kmap(v, half);
    a[2 * v]     = r[k];
    a[2 * v + 1] = r[k + 1];
  }
  return a;
}

// B[k,n] stored row-major (mem[k*ld + n]) : used for adj @ X (NN GEMM)
__device__ __forceinline__ v16h load_b_row(const f16* p, int ld) {
  int lane = threadIdx.x & 31;
  int col = lane & 15, half = lane >> 4;
  v16h b;
#pragma unroll
  for (int v = 0; v < 8; ++v) {
    int k = kmap(v, half);
    b[2 * v]     = p[(size_t)k * ld + col];
    b[2 * v + 1] = p[(size_t)(k + 1) * ld + col];
  }
  return b;
}

// B[k,n] where memory holds W[n,k] (mem[n*ld + k]) : used for X @ W^T
__device__ __forceinline__ v16h load_b_col(const f16* p, int ld) {
  int lane = threadIdx.x & 31;
  int col = lane & 15, half = lane >> 4;
  const f16* r = p + (size_t)col * ld;
  v16h b;
#pragma unroll
  for (int v = 0; v < 8; ++v) {
    int k = kmap(v, half);
    b[2 * v]     = r[k];
    b[2 * v + 1] = r[k + 1];
  }
  return b;
}

__device__ __forceinline__ v8f wmma16(v16h a, v16h b, v8f c) {
  return __builtin_amdgcn_wmma_f32_16x16x32_f16(false, a, false, b,
                                                (short)0, c, false, false);
}

__device__ __forceinline__ float sigm(float x) {
  return 1.0f / (1.0f + __expf(-x));
}

// ---- generic small kernels -------------------------------------------------
__global__ void conv_f16k(const float* __restrict__ s, f16* __restrict__ d, size_t n) {
  size_t i = (size_t)blockIdx.x * blockDim.x + threadIdx.x;
  size_t st = (size_t)gridDim.x * blockDim.x;
  for (; i < n; i += st) d[i] = (f16)s[i];
}

__global__ void zero_f32k(float* p, size_t n) {
  size_t i = (size_t)blockIdx.x * blockDim.x + threadIdx.x;
  size_t st = (size_t)gridDim.x * blockDim.x;
  for (; i < n; i += st) p[i] = 0.0f;
}

__global__ void copy_f16k(const f16* s, f16* d, size_t n) {
  size_t i = (size_t)blockIdx.x * blockDim.x + threadIdx.x;
  size_t st = (size_t)gridDim.x * blockDim.x;
  for (; i < n; i += st) d[i] = s[i];
}

// concat [Wih | Whh] rows -> f16, dst [4H, kih+khh]
__global__ void build_cat(const float* __restrict__ wih, const float* __restrict__ whh,
                          f16* __restrict__ dst, int kih, int khh) {
  int KC = kih + khh;
  int n = 4 * H_ * KC;
  int idx = blockIdx.x * blockDim.x + threadIdx.x;
  if (idx >= n) return;
  int r = idx / KC, c = idx % KC;
  float v = (c < kih) ? wih[(size_t)r * kih + c] : whh[(size_t)r * khh + (c - kih)];
  dst[idx] = (f16)v;
}

__global__ void bias_sumk(const float* a, const float* b, float* o, int n) {
  int i = blockIdx.x * blockDim.x + threadIdx.x;
  if (i < n) o[i] = a[i] + b[i];
}

// ---- GEMM: C[mr,n] = A[mr,K] @ W[n,K]^T (+bias) -> f16 --------------------
__global__ void gemm_nt_f16(const f16* __restrict__ A, const f16* __restrict__ W,
                            const float* __restrict__ bias, f16* __restrict__ C,
                            int N, int K) {
  int rt = blockIdx.x, ct = blockIdx.y;
  v8f acc[4] = {};
  const f16* Abase = A + (size_t)rt * 16 * K;
  for (int k = 0; k < K; k += 32) {
    v16h a = load_a16(Abase + k, K);
#pragma unroll
    for (int j = 0; j < 4; ++j) {
      v16h b = load_b_col(W + (size_t)(ct * 64 + j * 16) * K + k, K);
      acc[j] = wmma16(a, b, acc[j]);
    }
  }
  int lane = threadIdx.x & 31;
  int colL = lane & 15, half = lane >> 4;
#pragma unroll
  for (int j = 0; j < 4; ++j) {
    int col = ct * 64 + j * 16 + colL;
    float bv = bias ? bias[col] : 0.0f;
#pragma unroll
    for (int r = 0; r < 8; ++r) {
      int row = rt * 16 + (half ? r + 8 : r);
      C[(size_t)row * N + col] = (f16)(acc[j][r] + bv);
    }
  }
}

// ---- batched GEMM: C[b] = adj[b](MxM) @ X[b](MxN), optional relu ----------
__global__ void gemm_nn_batched(const f16* __restrict__ Adj, const f16* __restrict__ X,
                                f16* __restrict__ C, int N, int relu) {
  int rt = blockIdx.x, ct = blockIdx.y, bb = blockIdx.z;
  const f16* Ab = Adj + (size_t)bb * M_ * M_;
  const f16* Xb = X + (size_t)bb * M_ * N;
  f16* Cb = C + (size_t)bb * M_ * N;
  v8f acc[4] = {};
  for (int k = 0; k < M_; k += 32) {
    v16h a = load_a16(Ab + (size_t)rt * 16 * M_ + k, M_);
#pragma unroll
    for (int j = 0; j < 4; ++j) {
      v16h b = load_b_row(Xb + (size_t)k * N + ct * 64 + j * 16, N);
      acc[j] = wmma16(a, b, acc[j]);
    }
  }
  int lane = threadIdx.x & 31;
  int colL = lane & 15, half = lane >> 4;
#pragma unroll
  for (int j = 0; j < 4; ++j) {
    int col = ct * 64 + j * 16 + colL;
#pragma unroll
    for (int r = 0; r < 8; ++r) {
      int row = rt * 16 + (half ? r + 8 : r);
      float v = acc[j][r];
      if (relu && v < 0.0f) v = 0.0f;
      Cb[(size_t)row * N + col] = (f16)v;
    }
  }
}

// ---- persistent BiLSTM encoder: all 512 timesteps in one launch -----------
// grid (2 dirs, B/16 batch tiles), 512 threads = 16 waves (one per 16-col
// tile of H). h state in LDS, c state in accumulator-layout VGPRs, gate
// weight fragments hoisted into registers (loop-invariant over T).
__global__ void lstm_enc_scan(const f16* __restrict__ gnn,
                              const f16* __restrict__ wcat,
                              const float* __restrict__ bsum,
                              f16* __restrict__ enc) {
  const int KC = GO_ + H_;  // 320
  int d = blockIdx.x, rb = blockIdx.y;
  int wv = threadIdx.x >> 5;        // column tile 0..15
  int lane = threadIdx.x & 31;
  int colL = lane & 15, half = lane >> 4;
  int n = wv * 16 + colL;

  __shared__ f16 hsh[16 * H_];      // [row][n]  8 KB
  __shared__ f16 xsh[16 * GO_];     // [row][k]  2 KB

  const f16* wc = wcat + (size_t)d * 4 * H_ * KC;
  const float* bs = bsum + (size_t)d * 4 * H_;

  // hoist loop-invariant weight fragments: 10 K-chunks x 4 gates
  v16h bfrag[40];
#pragma unroll
  for (int it = 0; it < 10; ++it)
#pragma unroll
    for (int g = 0; g < 4; ++g)
      bfrag[it * 4 + g] =
          load_b_col(wc + (size_t)(g * H_ + wv * 16) * KC + it * 32, KC);

  float bb[4];
#pragma unroll
  for (int g = 0; g < 4; ++g) bb[g] = bs[g * H_ + n];

  for (int i = threadIdx.x; i < 16 * H_; i += 512) hsh[i] = (f16)0.0f;
  v8f creg = {};

  for (int t = 0; t < M_; ++t) {
    int t_idx = d ? (M_ - 1 - t) : t;
    // stage x_t tile (16 rows x 64) into LDS
    for (int i = threadIdx.x; i < 16 * GO_; i += 512) {
      int r = i >> 6, k = i & 63;
      xsh[i] = gnn[((size_t)(rb * 16 + r) * M_ + t_idx) * GO_ + k];
    }
    __syncthreads();  // xsh ready; previous-step hsh writes visible

    v8f acc[4] = {};
#pragma unroll
    for (int it = 0; it < 10; ++it) {
      int k = it * 32;
      v16h a = (k < GO_) ? load_a16(&xsh[k], GO_)
                         : load_a16(&hsh[k - GO_], H_);
#pragma unroll
      for (int g = 0; g < 4; ++g) acc[g] = wmma16(a, bfrag[it * 4 + g], acc[g]);
    }

    float hn[8];
#pragma unroll
    for (int r = 0; r < 8; ++r) {
      float iv = sigm(acc[0][r] + bb[0]);
      float fv = sigm(acc[1][r] + bb[1]);
      float gv = tanhf(acc[2][r] + bb[2]);
      float ov = sigm(acc[3][r] + bb[3]);
      float cn = fv * creg[r] + iv * gv;
      creg[r] = cn;
      hn[r] = ov * tanhf(cn);
    }
    __syncthreads();  // all waves done reading old hsh
#pragma unroll
    for (int r = 0; r < 8; ++r) {
      int rowL = half ? r + 8 : r;
      f16 hv = (f16)hn[r];
      hsh[rowL * H_ + n] = hv;
      enc[((size_t)(rb * 16 + rowL) * M_ + t_idx) * (2 * H_) + d * H_ + n] = hv;
    }
  }
}

// ---- decoder LSTM cell step: gates = [dec_in(512) | h(256)] @ Wcc^T -------
__global__ void lstm_dec_step(const f16* __restrict__ din, const f16* __restrict__ hsrc,
                              f16* __restrict__ hdst, float* __restrict__ cbuf,
                              const f16* __restrict__ wcc, const float* __restrict__ bs) {
  int rb = blockIdx.x, nc = blockIdx.y;
  const int KC = 3 * H_;  // 768
  v8f acc[4] = {};
  for (int it = 0; it < KC / 32; ++it) {
    int k = it * 32;
    v16h a = (k < 2 * H_)
                 ? load_a16(din + (size_t)rb * 16 * (2 * H_) + k, 2 * H_)
                 : load_a16(hsrc + (size_t)rb * 16 * H_ + (k - 2 * H_), H_);
#pragma unroll
    for (int g = 0; g < 4; ++g) {
      v16h b = load_b_col(wcc + (size_t)(g * H_ + nc * 16) * KC + k, KC);
      acc[g] = wmma16(a, b, acc[g]);
    }
  }
  int lane = threadIdx.x & 31;
  int n = nc * 16 + (lane & 15);
  int half = lane >> 4;
#pragma unroll
  for (int r = 0; r < 8; ++r) {
    int brow = rb * 16 + (half ? r + 8 : r);
    float iv = sigm(acc[0][r] + bs[0 * H_ + n]);
    float fv = sigm(acc[1][r] + bs[1 * H_ + n]);
    float gv = tanhf(acc[2][r] + bs[2 * H_ + n]);
    float ov = sigm(acc[3][r] + bs[3 * H_ + n]);
    float cn = fv * cbuf[(size_t)brow * H_ + n] + iv * gv;
    float hnv = ov * tanhf(cn);
    cbuf[(size_t)brow * H_ + n] = cn;
    hdst[(size_t)brow * H_ + n] = (f16)hnv;
  }
}

// ---- enc mean over M: [B,2H] f32 + f16 ------------------------------------
__global__ void enc_mean_kernel(const f16* __restrict__ enc, float* __restrict__ mf,
                                f16* __restrict__ mh) {
  int idx = blockIdx.x * blockDim.x + threadIdx.x;
  if (idx >= B_ * 2 * H_) return;
  int b = idx / (2 * H_), dd = idx % (2 * H_);
  float s = 0.0f;
  for (int m = 0; m < M_; ++m) s += (float)enc[((size_t)b * M_ + m) * (2 * H_) + dd];
  s *= (1.0f / (float)M_);
  mf[idx] = s;
  mh[idx] = (f16)s;
}

// ---- h0/c0 = enc_mean @ initW^T + b ---------------------------------------
__global__ void init_hc(const float* __restrict__ mean, const float* __restrict__ Wh,
                        const float* __restrict__ bh, const float* __restrict__ Wc,
                        const float* __restrict__ bc, f16* __restrict__ h0,
                        float* __restrict__ c0) {
  int idx = blockIdx.x * blockDim.x + threadIdx.x;
  if (idx >= B_ * H_) return;
  int b = idx / H_, n = idx % H_;
  const float* mr = mean + (size_t)b * 2 * H_;
  const float* whr = Wh + (size_t)n * 2 * H_;
  const float* wcr = Wc + (size_t)n * 2 * H_;
  float sh = bh[n], sc = bc[n];
  for (int dd = 0; dd < 2 * H_; ++dd) {
    float e = mr[dd];
    sh += e * whr[dd];
    sc += e * wcr[dd];
  }
  h0[idx] = (f16)sh;
  c0[idx] = sc;
}

// ---- query = h @ ptr_W^T + ptr_b -> f16 [B, 2H] ---------------------------
__global__ void query_kernel(const f16* __restrict__ h, const float* __restrict__ W,
                             const float* __restrict__ bias, f16* __restrict__ q) {
  int idx = blockIdx.x * blockDim.x + threadIdx.x;
  if (idx >= B_ * 2 * H_) return;
  int b = idx / (2 * H_), qq = idx % (2 * H_);
  const f16* hr = h + (size_t)b * H_;
  const float* wr = W + (size_t)qq * H_;
  float s = bias[qq];
  for (int n = 0; n < H_; ++n) s += (float)hr[n] * wr[n];
  q[idx] = (f16)s;
}

// ---- scores[b,m] = enc[b,m,:] . query[b,:], masked ------------------------
__global__ void scores_kernel(const f16* __restrict__ enc, const f16* __restrict__ q,
                              const float* __restrict__ mask, float* __restrict__ sc) {
  int idx = blockIdx.x * blockDim.x + threadIdx.x;
  if (idx >= B_ * M_) return;
  int b = idx / M_;
  const f16* er = enc + (size_t)idx * (2 * H_);
  const f16* qr = q + (size_t)b * (2 * H_);
  float s = 0.0f;
  for (int dd = 0; dd < 2 * H_; ++dd) s += (float)er[dd] * (float)qr[dd];
  sc[idx] = (mask[idx] > 0.5f) ? NEGV : s;
}

// ---- masked-softmax inverse-CDF sample + logp + gather dec_in -------------
__global__ void sample_step(int step, const float* __restrict__ sc,
                            float* __restrict__ mask, const f16* __restrict__ enc,
                            f16* __restrict__ din, float* __restrict__ logacc,
                            float* __restrict__ out_idx, float* __restrict__ out_lp) {
  int b = blockIdx.x;
  int tid = threadIdx.x;
  __shared__ float red[256];
  __shared__ float s_max;
  __shared__ int s_sel;
  const float* sr = sc + (size_t)b * M_;

  red[tid] = fmaxf(sr[tid], sr[tid + 256]);
  __syncthreads();
  for (int off = 128; off > 0; off >>= 1) {
    if (tid < off) red[tid] = fmaxf(red[tid], red[tid + off]);
    __syncthreads();
  }
  if (tid == 0) s_max = red[0];
  __syncthreads();
  float mx = s_max;
  red[tid] = __expf(sr[tid] - mx) + __expf(sr[tid + 256] - mx);
  __syncthreads();
  for (int off = 128; off > 0; off >>= 1) {
    if (tid < off) red[tid] += red[tid + off];
    __syncthreads();
  }
  if (tid == 0) {
    float tot = red[0];
    unsigned x = ((unsigned)step * 0x9E3779B1u) ^ ((unsigned)b + 0x7F4A7C15u);
    x ^= x >> 16; x *= 0x85EBCA6Bu; x ^= x >> 13; x *= 0xC2B2AE35u; x ^= x >> 16;
    float u = (float)(x >> 8) * (1.0f / 16777216.0f) * tot;
    float cum = 0.0f;
    int sel = M_ - 1, found = 0;
    for (int m = 0; m < M_; ++m) {
      cum += __expf(sr[m] - mx);
      if (!found && cum >= u) { sel = m; found = 1; }
    }
    float lp = sr[sel] - mx - logf(tot);
    float acc = logacc[b] + lp;
    logacc[b] = acc;
    out_idx[(size_t)b * NSEL + step] = (float)sel;
    out_lp[b] = acc;
    mask[(size_t)b * M_ + sel] = 1.0f;
    s_sel = sel;
  }
  __syncthreads();
  int sel = s_sel;
  for (int dd = tid; dd < 2 * H_; dd += 256)
    din[(size_t)b * (2 * H_) + dd] = enc[((size_t)b * M_ + sel) * (2 * H_) + dd];
}

// ---------------------------------------------------------------------------
extern "C" void kernel_launch(void* const* d_in, const int* in_sizes, int n_in,
                              void* d_out, int out_size, void* d_ws, size_t ws_size,
                              hipStream_t stream) {
  (void)in_sizes; (void)n_in; (void)out_size; (void)ws_size;
  const float* inputs = (const float*)d_in[0];
  const float* adj    = (const float*)d_in[1];
  const float* gc1_W  = (const float*)d_in[2];
  const float* gc1_b  = (const float*)d_in[3];
  const float* gc2_W  = (const float*)d_in[4];
  const float* gc2_b  = (const float*)d_in[5];
  const float* Wih_f  = (const float*)d_in[6];
  const float* Whh_f  = (const float*)d_in[7];
  const float* bih_f  = (const float*)d_in[8];
  const float* bhh_f  = (const float*)d_in[9];
  const float* Wih_b  = (const float*)d_in[10];
  const float* Whh_b  = (const float*)d_in[11];
  const float* bih_b  = (const float*)d_in[12];
  const float* bhh_b  = (const float*)d_in[13];
  const float* initW_h = (const float*)d_in[14];
  const float* initb_h = (const float*)d_in[15];
  const float* initW_c = (const float*)d_in[16];
  const float* initb_c = (const float*)d_in[17];
  const float* cWih   = (const float*)d_in[18];
  const float* cWhh   = (const float*)d_in[19];
  const float* cbih   = (const float*)d_in[20];
  const float* cbhh   = (const float*)d_in[21];
  const float* ptrW   = (const float*)d_in[22];
  const float* ptrb   = (const float*)d_in[23];
  float* out = (float*)d_out;

  char* ws = (char*)d_ws;
  size_t off = 0;
  auto alloc = [&](size_t bytes) -> void* {
    void* p = ws + off;
    off += (bytes + 255) & ~(size_t)255;
    return p;
  };

  f16*   adj_h  = (f16*)alloc((size_t)B_ * M_ * M_ * 2);
  f16*   in_h   = (f16*)alloc((size_t)B_ * M_ * D_ * 2);
  f16*   y1_h   = (f16*)alloc((size_t)B_ * M_ * GH_ * 2);
  f16*   x_h    = (f16*)alloc((size_t)B_ * M_ * GH_ * 2);
  f16*   y2_h   = (f16*)alloc((size_t)B_ * M_ * GO_ * 2);
  f16*   gnn_h  = (f16*)alloc((size_t)B_ * M_ * GO_ * 2);
  f16*   enc_h  = (f16*)alloc((size_t)B_ * M_ * 2 * H_ * 2);
  f16*   gc1W_h = (f16*)alloc((size_t)GH_ * D_ * 2);
  f16*   gc2W_h = (f16*)alloc((size_t)GO_ * GH_ * 2);
  f16*   wcat   = (f16*)alloc((size_t)2 * 4 * H_ * (GO_ + H_) * 2);
  float* bsum   = (float*)alloc((size_t)2 * 4 * H_ * 4);
  f16*   wcc    = (f16*)alloc((size_t)4 * H_ * 3 * H_ * 2);
  float* bsumc  = (float*)alloc((size_t)4 * H_ * 4);
  float* meanf  = (float*)alloc((size_t)B_ * 2 * H_ * 4);
  f16*   meanh  = (f16*)alloc((size_t)B_ * 2 * H_ * 2);
  f16*   hdec   = (f16*)alloc((size_t)2 * B_ * H_ * 2);      // ping-pong
  float* cdec   = (float*)alloc((size_t)B_ * H_ * 4);
  f16*   din    = (f16*)alloc((size_t)B_ * 2 * H_ * 2);
  f16*   q16    = (f16*)alloc((size_t)B_ * 2 * H_ * 2);
  float* scb    = (float*)alloc((size_t)B_ * M_ * 4);
  float* maskb  = (float*)alloc((size_t)B_ * M_ * 4);
  float* logacc = (float*)alloc((size_t)B_ * 4);

  // ---- f32 -> f16 conversions + packed weights ----
  conv_f16k<<<8192, 256, 0, stream>>>(adj, adj_h, (size_t)B_ * M_ * M_);
  conv_f16k<<<4096, 256, 0, stream>>>(inputs, in_h, (size_t)B_ * M_ * D_);
  conv_f16k<<<64, 256, 0, stream>>>(gc1_W, gc1W_h, (size_t)GH_ * D_);
  conv_f16k<<<64, 256, 0, stream>>>(gc2_W, gc2W_h, (size_t)GO_ * GH_);
  {
    int n = 4 * H_ * (GO_ + H_);
    build_cat<<<(n + 255) / 256, 256, 0, stream>>>(Wih_f, Whh_f, wcat, GO_, H_);
    build_cat<<<(n + 255) / 256, 256, 0, stream>>>(Wih_b, Whh_b, wcat + (size_t)n, GO_, H_);
    int nc = 4 * H_ * 3 * H_;
    build_cat<<<(nc + 255) / 256, 256, 0, stream>>>(cWih, cWhh, wcc, 2 * H_, H_);
  }
  bias_sumk<<<4, 256, 0, stream>>>(bih_f, bhh_f, bsum, 4 * H_);
  bias_sumk<<<4, 256, 0, stream>>>(bih_b, bhh_b, bsum + 4 * H_, 4 * H_);
  bias_sumk<<<4, 256, 0, stream>>>(cbih, cbhh, bsumc, 4 * H_);

  // ---- GNN: relu(adj @ (in @ gc1^T + b)) -> adj @ (x @ gc2^T + b) ----
  gemm_nt_f16<<<dim3(B_ * M_ / 16, GH_ / 64), 32, 0, stream>>>(in_h, gc1W_h, gc1_b, y1_h, GH_, D_);
  gemm_nn_batched<<<dim3(M_ / 16, GH_ / 64, B_), 32, 0, stream>>>(adj_h, y1_h, x_h, GH_, 1);
  gemm_nt_f16<<<dim3(B_ * M_ / 16, GO_ / 64), 32, 0, stream>>>(x_h, gc2W_h, gc2_b, y2_h, GO_, GH_);
  gemm_nn_batched<<<dim3(M_ / 16, GO_ / 64, B_), 32, 0, stream>>>(adj_h, y2_h, gnn_h, GO_, 0);

  // ---- bidirectional LSTM encoder: one persistent kernel, 512 steps ----
  lstm_enc_scan<<<dim3(2, B_ / 16), 512, 0, stream>>>(gnn_h, wcat, bsum, enc_h);

  // ---- decoder init ----
  enc_mean_kernel<<<(B_ * 2 * H_ + 255) / 256, 256, 0, stream>>>(enc_h, meanf, meanh);
  init_hc<<<(B_ * H_ + 255) / 256, 256, 0, stream>>>(meanf, initW_h, initb_h,
                                                     initW_c, initb_c, hdec, cdec);
  copy_f16k<<<256, 256, 0, stream>>>(meanh, din, (size_t)B_ * 2 * H_);
  zero_f32k<<<256, 256, 0, stream>>>(maskb, (size_t)B_ * M_);
  zero_f32k<<<1, 256, 0, stream>>>(logacc, (size_t)B_);

  // ---- 32 pointer decode steps ----
  for (int s = 0; s < NSEL; ++s) {
    f16* hsrc = hdec + (size_t)(s & 1) * B_ * H_;
    f16* hdst = hdec + (size_t)((s + 1) & 1) * B_ * H_;
    lstm_dec_step<<<dim3(B_ / 16, H_ / 16), 32, 0, stream>>>(din, hsrc, hdst, cdec, wcc, bsumc);
    query_kernel<<<(B_ * 2 * H_ + 255) / 256, 256, 0, stream>>>(hdst, ptrW, ptrb, q16);
    scores_kernel<<<(B_ * M_ + 255) / 256, 256, 0, stream>>>(enc_h, q16, maskb, scb);
    sample_step<<<B_, 256, 0, stream>>>(s, scb, maskb, enc_h, din, logacc,
                                        out, out + (size_t)B_ * NSEL);
  }
}
